// VectorQuantizer_36739150250690
// MI455X (gfx1250) — matrix operand order repaired and checked by reading, not compiled, not run on hardware
//
#include <hip/hip_runtime.h>
#include <hip/hip_bf16.h>

typedef __attribute__((ext_vector_type(16))) __bf16 v16bf;
typedef __attribute__((ext_vector_type(8)))  __bf16 v8bf;
typedef __attribute__((ext_vector_type(8)))  float  v8f;

#define VQ_HW   4096      // 64*64
#define VQ_C    256       // channels / K-dim of GEMM
#define VQ_K    1024      // codebook size
#define VQ_B    16        // batch
#define ROWS_PER_BLOCK 128
#define LDS_STRIDE 264    // 256 + 8 bf16 pad (avoid LDS bank pile-up)
#define SMEM_BYTES (ROWS_PER_BLOCK * LDS_STRIDE * 2 * 2)  // hi+lo, 2B each = 135168

// ---------------------------------------------------------------------------
// Kernel 1: codebook prep — e2[k] = ||E_k||^2, and bf16 hi/lo split copies.
// ---------------------------------------------------------------------------
__global__ void vq_prep(const float* __restrict__ emb,
                        float* __restrict__ e2,
                        __bf16* __restrict__ ehi,
                        __bf16* __restrict__ elo) {
    const int code = blockIdx.x;   // 1024 blocks
    const int c    = threadIdx.x;  // 256 threads
    const float v  = emb[code * VQ_C + c];
    const __bf16 h = (__bf16)v;
    ehi[code * VQ_C + c] = h;
    elo[code * VQ_C + c] = (__bf16)(v - (float)h);

    __shared__ float red[256];
    red[c] = v * v;
    __syncthreads();
    for (int s = 128; s > 0; s >>= 1) {
        if (c < s) red[c] += red[c + s];
        __syncthreads();
    }
    if (c == 0) e2[code] = red[0];
}

// ---------------------------------------------------------------------------
// Kernel 2: WMMA GEMM (bf16x3 split) + row-wise argmin over 1024 codes.
// Block = 8 waves x 16 rows = 128 rows of xf. x-tile staged in LDS as bf16.
// score(row, code) = e2[code] - 2 * <x_row, E_code>   (||x||^2 constant/row)
// ---------------------------------------------------------------------------
__global__ __launch_bounds__(256, 1) void vq_argmin(
    const float*  __restrict__ x,
    const __bf16* __restrict__ ehi,
    const __bf16* __restrict__ elo,
    const float*  __restrict__ e2,
    int*          __restrict__ idx) {
    extern __shared__ char smem[];
    __bf16* ahiLds = (__bf16*)smem;                               // [128][264]
    __bf16* aloLds = ahiLds + ROWS_PER_BLOCK * LDS_STRIDE;        // [128][264]

    const int tid     = threadIdx.x;
    const int lane    = tid & 31;
    const int wave    = tid >> 5;
    const int rowbase = blockIdx.x * ROWS_PER_BLOCK;   // global row (b*4096 + n)
    const int bIdx    = rowbase >> 12;                 // rowbase / 4096
    const int n0      = rowbase & (VQ_HW - 1);
    const float* xb   = x + (size_t)bIdx * VQ_C * VQ_HW;

    // ---- Stage 128 rows x 256 channels into LDS (bf16 hi/lo), coalesced ----
    for (int i = 0; i < (ROWS_PER_BLOCK * VQ_C) / 256; ++i) {
        const int e  = i * 256 + tid;
        const int c  = e >> 7;       // channel
        const int nf = e & 127;      // row within tile
        const float v = xb[(size_t)c * VQ_HW + (n0 + nf)];
        const __bf16 h = (__bf16)v;
        ahiLds[nf * LDS_STRIDE + c] = h;
        aloLds[nf * LDS_STRIDE + c] = (__bf16)(v - (float)h);
    }
    __syncthreads();

    // A fragment (16-bit A 16x32): lanes 0-15 hold K 0..15,
    // lanes 16-31 hold K 16..31 for the same row M = lane&15 (matches B).
    const int nloc  = wave * 16 + (lane & 15);     // row within block tile
    const int aHalf = (lane < 16) ? 0 : 16;        // A K-offset inside 32-chunk
    // B fragment (32x16): lanes 0-15 hold K 0..15, lanes 16-31 K 16..31, N=lane&15.
    const int kHalf = (lane < 16) ? 0 : 16;

    float bestVal[8];
    int   bestIdx[8];
#pragma unroll
    for (int j = 0; j < 8; ++j) { bestVal[j] = 3.4e38f; bestIdx[j] = 0; }

    for (int ct = 0; ct < VQ_K / 16; ++ct) {       // 64 code tiles
        const int code = ct * 16 + (lane & 15);    // this lane's column N
        v8f acc = {};
#pragma unroll
        for (int kc = 0; kc < VQ_C / 32; ++kc) {   // 8 K-chunks of 32
            const int cb = kc * 32 + aHalf;
            v16bf afHi, afLo;
            {
                const v8bf p0 = *(const v8bf*)&ahiLds[nloc * LDS_STRIDE + cb];
                const v8bf p1 = *(const v8bf*)&ahiLds[nloc * LDS_STRIDE + cb + 8];
                const v8bf q0 = *(const v8bf*)&aloLds[nloc * LDS_STRIDE + cb];
                const v8bf q1 = *(const v8bf*)&aloLds[nloc * LDS_STRIDE + cb + 8];
#pragma unroll
                for (int e = 0; e < 8; ++e) {
                    afHi[e] = p0[e]; afHi[8 + e] = p1[e];
                    afLo[e] = q0[e]; afLo[8 + e] = q1[e];
                }
            }
            const size_t boff = (size_t)code * VQ_C + kc * 32 + kHalf;
            const v16bf bfHi = *(const v16bf*)(ehi + boff);   // 32B contiguous
            const v16bf bfLo = *(const v16bf*)(elo + boff);
            // bf16x3 error compensation: hi*hi + hi*lo + lo*hi (f32 accumulate)
            acc = __builtin_amdgcn_wmma_f32_16x16x32_bf16(false, afHi, false, bfHi,
                                                          (short)0, acc, false, false);
            acc = __builtin_amdgcn_wmma_f32_16x16x32_bf16(false, afHi, false, bfLo,
                                                          (short)0, acc, false, false);
            acc = __builtin_amdgcn_wmma_f32_16x16x32_bf16(false, afLo, false, bfHi,
                                                          (short)0, acc, false, false);
        }
        const float e2c = e2[code];
        // C layout: lanes 0-15 VGPR j -> row j; lanes 16-31 VGPR j -> row j+8.
#pragma unroll
        for (int j = 0; j < 8; ++j) {
            const float s = fmaf(-2.0f, acc[j], e2c);
            if (s < bestVal[j]) { bestVal[j] = s; bestIdx[j] = code; }
        }
    }

    // Cross-lane argmin within each 16-lane half (tie-break: lowest index,
    // matching jnp.argmin first-occurrence semantics).
#pragma unroll
    for (int j = 0; j < 8; ++j) {
        float bv = bestVal[j];
        int   bi = bestIdx[j];
#pragma unroll
        for (int off = 1; off < 16; off <<= 1) {
            const float ov = __shfl_xor(bv, off, 32);
            const int   oi = __shfl_xor(bi, off, 32);
            if (ov < bv || (ov == bv && oi < bi)) { bv = ov; bi = oi; }
        }
        bestIdx[j] = bi;
    }
    if ((lane & 15) == 0) {
        const int rowOff = (lane < 16) ? 0 : 8;
#pragma unroll
        for (int j = 0; j < 8; ++j)
            idx[rowbase + wave * 16 + rowOff + j] = bestIdx[j];
    }
}

// ---------------------------------------------------------------------------
// Kernel 3: gather winning f32 rows into (b,c,h,w) layout + loss partials.
// grid = (16 n-tiles, 16 batches), 256 threads; thread owns one n.
// ---------------------------------------------------------------------------
__global__ void vq_gather(const float* __restrict__ x,
                          const float* __restrict__ emb,
                          const int*   __restrict__ idx,
                          float*       __restrict__ out,
                          float*       __restrict__ partials) {
    const int bIdx = blockIdx.y;
    const int n    = blockIdx.x * 256 + threadIdx.x;
    const int code = idx[bIdx * VQ_HW + n];
    const float* er   = emb + (size_t)code * VQ_C;
    const float* xr   = x   + (size_t)bIdx * VQ_C * VQ_HW + n;
    float*       orow = out + (size_t)bIdx * VQ_C * VQ_HW + n;

    float acc = 0.0f;
#pragma unroll 4
    for (int c = 0; c < VQ_C; ++c) {
        const float ev = er[c];
        const float xv = xr[(size_t)c * VQ_HW];
        orow[(size_t)c * VQ_HW] = ev;              // coalesced over n
        const float d = ev - xv;
        acc = fmaf(d, d, acc);
    }

    __shared__ float red[256];
    red[threadIdx.x] = acc;
    __syncthreads();
    for (int s = 128; s > 0; s >>= 1) {
        if (threadIdx.x < s) red[threadIdx.x] += red[threadIdx.x + s];
        __syncthreads();
    }
    if (threadIdx.x == 0) partials[bIdx * 16 + blockIdx.x] = red[0];
}

// ---------------------------------------------------------------------------
// Kernel 4: deterministic fixed-order final reduction of 256 partials.
// loss = 2 * sum / (b*hw*c)   (both stop-gradient terms are equal numerically)
// ---------------------------------------------------------------------------
__global__ void vq_loss(const float* __restrict__ partials,
                        float* __restrict__ loss) {
    __shared__ float red[256];
    red[threadIdx.x] = partials[threadIdx.x];
    __syncthreads();
    for (int s = 128; s > 0; s >>= 1) {
        if (threadIdx.x < s) red[threadIdx.x] += red[threadIdx.x + s];
        __syncthreads();
    }
    if (threadIdx.x == 0) loss[0] = 2.0f * red[0] / 16777216.0f;
}

// ---------------------------------------------------------------------------
extern "C" void kernel_launch(void* const* d_in, const int* in_sizes, int n_in,
                              void* d_out, int out_size, void* d_ws, size_t ws_size,
                              hipStream_t stream) {
    (void)in_sizes; (void)n_in; (void)out_size; (void)ws_size;
    const float* x   = (const float*)d_in[0];   // (16,256,64,64) f32
    const float* emb = (const float*)d_in[1];   // (1024,256) f32
    float* out  = (float*)d_out;                // z_q_img: 16777216 f32
    float* loss = out + (size_t)VQ_B * VQ_C * VQ_HW;  // + scalar

    // workspace carve (hipMalloc base is 4KB-aligned; all offsets 4KB multiples)
    char* ws = (char*)d_ws;
    float*  e2       = (float*)ws;                            //   4 KB
    __bf16* ehi      = (__bf16*)(ws + 4096);                  // 512 KB
    __bf16* elo      = (__bf16*)(ws + 4096 + 524288);         // 512 KB
    int*    idx      = (int*)(ws + 4096 + 2 * 524288);        // 256 KB
    float*  partials = (float*)(ws + 4096 + 2 * 524288 + 262144); // 1 KB

    // opt in to 132KB dynamic LDS (CDNA5 WGP has 320KB)
    (void)hipFuncSetAttribute((const void*)vq_argmin,
                              hipFuncAttributeMaxDynamicSharedMemorySize,
                              SMEM_BYTES);

    vq_prep  <<<VQ_K, 256, 0, stream>>>(emb, e2, ehi, elo);
    vq_argmin<<<(VQ_B * VQ_HW) / ROWS_PER_BLOCK, 256, SMEM_BYTES, stream>>>(
        x, ehi, elo, e2, idx);
    vq_gather<<<dim3(VQ_HW / 256, VQ_B), 256, 0, stream>>>(x, emb, idx, out, partials);
    vq_loss  <<<1, 256, 0, stream>>>(partials, loss);
}
